// Model_18820546691263
// MI455X (gfx1250) — matrix operand order repaired and checked by reading, not compiled, not run on hardware
//
#include <hip/hip_runtime.h>
#include <math.h>

typedef __bf16 bf16;
typedef unsigned int u32;
typedef __attribute__((ext_vector_type(16))) __bf16 v16bf;
typedef __attribute__((ext_vector_type(8)))  float   v8f;

#define DEV __device__ __forceinline__

static constexpr int Dm = 128;
static constexpr float ATT_SCALE = 0.08838834764831845f; // 128^-0.5

union Frag {
    v16bf v;
    u32   d[8];
};

DEV u32 b2u(bf16 x) {
    unsigned short s = __builtin_bit_cast(unsigned short, x);
    return (u32)s;
}

// ---------------- elementwise / conv / bn kernels ----------------

__global__ void k_f32_to_bf16(const float* __restrict__ in, bf16* __restrict__ out, int n) {
    int i = blockIdx.x * blockDim.x + threadIdx.x;
    if (i < n) out[i] = (bf16)in[i];
}

__global__ void k_conv1d(const float* __restrict__ x, const float* __restrict__ w,
                         const float* __restrict__ bias, float* __restrict__ y,
                         int B, int Cin, int Cout, int T, int K) {
    int idx = blockIdx.x * blockDim.x + threadIdx.x;
    int total = B * Cout * T;
    if (idx >= total) return;
    int t  = idx % T;
    int oc = (idx / T) % Cout;
    int b  = idx / (T * Cout);
    int P = K / 2;
    float acc = bias[oc];
    const float* xb = x + (size_t)b * Cin * T;
    const float* wo = w + (size_t)oc * Cin * K;
    for (int ic = 0; ic < Cin; ++ic) {
        const float* xc = xb + (size_t)ic * T;
        const float* wc = wo + (size_t)ic * K;
        for (int k = 0; k < K; ++k) {
            int tt = t - P + k;
            if (tt >= 0 && tt < T) acc += xc[tt] * wc[k];
        }
    }
    y[idx] = acc;
}

__global__ void k_bnstats(const float* __restrict__ x, float* __restrict__ stats,
                          int B, int C, int T) {
    int c = blockIdx.x;
    __shared__ float ssum[256];
    __shared__ float ssq[256];
    float s = 0.f, q = 0.f;
    int n = B * T;
    for (int i = threadIdx.x; i < n; i += blockDim.x) {
        int b = i / T, t = i % T;
        float v = x[((size_t)b * C + c) * T + t];
        s += v; q += v * v;
    }
    ssum[threadIdx.x] = s; ssq[threadIdx.x] = q;
    __syncthreads();
    for (int st = blockDim.x / 2; st > 0; st >>= 1) {
        if ((int)threadIdx.x < st) {
            ssum[threadIdx.x] += ssum[threadIdx.x + st];
            ssq[threadIdx.x]  += ssq[threadIdx.x + st];
        }
        __syncthreads();
    }
    if (threadIdx.x == 0) {
        float mean = ssum[0] / (float)n;
        float var  = ssq[0] / (float)n - mean * mean;
        stats[c]     = mean;
        stats[C + c] = rsqrtf(var + 1e-5f);
    }
}

__global__ void k_bn_relu(float* __restrict__ x, const float* __restrict__ stats,
                          const float* __restrict__ g, const float* __restrict__ bta,
                          int B, int C, int T) {
    int idx = blockIdx.x * blockDim.x + threadIdx.x;
    int total = B * C * T;
    if (idx >= total) return;
    int c = (idx / T) % C;
    float v = (x[idx] - stats[c]) * stats[C + c] * g[c] + bta[c];
    x[idx] = v > 0.f ? v : 0.f;
}

__global__ void k_demean(const float* __restrict__ x, float* __restrict__ y, int BC, int T) {
    int bc = blockIdx.x * blockDim.x + threadIdx.x;
    if (bc >= BC) return;
    const float* xr = x + (size_t)bc * T;
    float* yr = y + (size_t)bc * T;
    float s = 0.f;
    for (int t = 0; t < T; ++t) s += xr[t];
    s /= (float)T;
    for (int t = 0; t < T; ++t) yr[t] = xr[t] - s;
}

// [B,C,T] f32 -> [(B*T), C] bf16
__global__ void k_tc(const float* __restrict__ x, bf16* __restrict__ y, int B, int C, int T) {
    int idx = blockIdx.x * blockDim.x + threadIdx.x;
    int total = B * T * C;
    if (idx >= total) return;
    int c  = idx % C;
    int bt = idx / C;
    int t  = bt % T;
    int b  = bt / T;
    y[idx] = (bf16)x[((size_t)b * C + c) * T + t];
}

// Pack a [Bt, Lk, 128] bf16 matrix into WMMA B-fragment layout:
// dst[((batch*8 + nt)*KC + kc)*256 + lane*8 + i] packs rows k0=kc*32+16*(i>>2)+8*(lane>>4)+2*(i&3),
// k0+1 at column n = nt*16 + (lane&15). Rows >= Lk are zero-padded.
__global__ void k_packB(const bf16* __restrict__ src, u32* __restrict__ dst,
                        int Lk, int KC, int Bt) {
    int idx = blockIdx.x * blockDim.x + threadIdx.x;
    int total = Bt * 8 * KC * 256;
    if (idx >= total) return;
    int i    = idx & 7;
    int lane = (idx >> 3) & 31;
    int blk  = idx >> 8;
    int kc   = blk % KC;
    int nt   = (blk / KC) & 7;
    int batch = blk / (KC * 8);
    int nn = lane & 15, hh = lane >> 4;
    int k0 = kc * 32 + 16 * (i >> 2) + 8 * hh + 2 * (i & 3);
    int n  = nt * 16 + nn;
    const bf16* sb = src + (size_t)batch * Lk * 128;
    u32 lo = (k0     < Lk) ? b2u(sb[(size_t)k0 * 128 + n])       : 0u;
    u32 hi = (k0 + 1 < Lk) ? b2u(sb[(size_t)(k0 + 1) * 128 + n]) : 0u;
    dst[idx] = lo | (hi << 16);
}

// ---------------- WMMA GEMM: A[M,K] bf16 RM * Wpk(packed)[K,128] + bias -> bf16 [M,128]
// one wave computes a 16x64 output slab (4 N-tiles reuse one A fragment)

__global__ void k_gemm128(const bf16* __restrict__ A, const u32* __restrict__ Wpk,
                          const float* __restrict__ bias, bf16* __restrict__ outb,
                          int M, int K) {
    int lane  = threadIdx.x & 31;
    int wid   = threadIdx.x >> 5;
    int wtile = blockIdx.x * (blockDim.x >> 5) + wid;
    int total = (M >> 4) * 2; // 2 column groups of 64
    if (wtile >= total) return;
    int tm = wtile >> 1, tg = wtile & 1;
    int KC = K >> 5;
    int nn = lane & 15, h = lane >> 4;
    const u32* arow = (const u32*)(A + (size_t)tm * 16 * K + (size_t)nn * K);
    v8f acc[4] = {};
    for (int kc = 0; kc < KC; ++kc) {
        Frag a;
#pragma unroll
        for (int i = 0; i < 4; ++i) a.d[i]     = arow[kc * 16 + 4 * h + i];
#pragma unroll
        for (int i = 0; i < 4; ++i) a.d[4 + i] = arow[kc * 16 + 8 + 4 * h + i];
#pragma unroll
        for (int nt = 0; nt < 4; ++nt) {
            const u32* wp = Wpk + ((size_t)(tg * 4 + nt) * KC + kc) * 256 + lane * 8;
            Frag b;
#pragma unroll
            for (int i = 0; i < 8; ++i) b.d[i] = wp[i];
            acc[nt] = __builtin_amdgcn_wmma_f32_16x16x32_bf16(false, a.v, false, b.v,
                                                              (short)0, acc[nt], false, false);
        }
    }
#pragma unroll
    for (int nt = 0; nt < 4; ++nt) {
        int n = (tg * 4 + nt) * 16 + nn;
        float bv = bias[n];
#pragma unroll
        for (int i = 0; i < 8; ++i) {
            int m = tm * 16 + i + 8 * h;
            outb[(size_t)m * 128 + n] = (bf16)(acc[nt][i] + bv);
        }
    }
}

// ---------------- flash-style banded cross attention ----------------
// one wave per (batch, 16-row Q tile). D=128. scores in regs, probs via 1KB LDS,
// V pre-packed in B-fragment layout (zero padded past Lk).
__global__ void k_attn(const bf16* __restrict__ Q, const bf16* __restrict__ Kmat,
                       const u32* __restrict__ Vpk, float* __restrict__ Out,
                       int Lq, int Lk, int maskmode) {
    int tilesQ = (Lq + 15) >> 4;
    int batch = blockIdx.x / tilesQ;
    int tq    = blockIdx.x % tilesQ;
    int lane = threadIdx.x;
    int nn = lane & 15, h = lane >> 4;
    int KC = (Lk + 31) >> 5;
    const bf16* Qb  = Q    + (size_t)batch * Lq * Dm;
    const bf16* Kb  = Kmat + (size_t)batch * Lk * Dm;
    const u32*  Vpb = Vpk  + (size_t)batch * 8 * KC * 256;
    float* Ob = Out + (size_t)batch * Lq * Dm;
    int r0 = tq * 16;

    // Q A-fragments, 4 chunks of K=32 covering D=128 (dword loads, contiguous pairs)
    Frag qf[4];
    {
        int rq = r0 + nn;
        if (rq >= Lq) rq = Lq - 1;
        const u32* row = (const u32*)(Qb + (size_t)rq * Dm);
#pragma unroll
        for (int c = 0; c < 4; ++c) {
#pragma unroll
            for (int i = 0; i < 4; ++i) qf[c].d[i]     = row[c * 16 + 4 * h + i];
#pragma unroll
            for (int i = 0; i < 4; ++i) qf[c].d[4 + i] = row[c * 16 + 8 + 4 * h + i];
        }
    }

    v8f acc[8] = {};
    float rm[8], rs[8];
#pragma unroll
    for (int i = 0; i < 8; ++i) { rm[i] = -3.0e38f; rs[i] = 0.f; }

    __shared__ __align__(16) bf16 ptile[16 * 32];
    u32* pt32 = (u32*)ptile;

    int nchunks = KC;
    for (int cc = 0; cc < nchunks; ++cc) {
        int c0 = cc * 32;
        float sv[2][8];
#pragma unroll
        for (int ct = 0; ct < 2; ++ct) {
            int col = c0 + ct * 16 + nn;
            int colc = col < Lk ? col : Lk - 1;
            const u32* krow = (const u32*)(Kb + (size_t)colc * Dm);
            v8f s = {};
#pragma unroll
            for (int kc = 0; kc < 4; ++kc) {
                Frag bf_;
#pragma unroll
                for (int i = 0; i < 8; ++i) bf_.d[i] = krow[kc * 16 + 8 * (i >> 2) + 4 * h + (i & 3)];
                s = __builtin_amdgcn_wmma_f32_16x16x32_bf16(false, qf[kc].v, false, bf_.v,
                                                            (short)0, s, false, false);
            }
#pragma unroll
            for (int i = 0; i < 8; ++i) {
                int r = r0 + i + 8 * h;
                float v = s[i] * ATT_SCALE;
                bool ok;
                if (maskmode == 0) { int j0 = r / 20;   ok = (col >= j0 + 20) && (col <= j0 + 80); }
                else               { int j0 = col / 20; ok = (r   >= j0 + 20) && (r   <= j0 + 80); }
                if (col >= Lk)      v = -3.0e38f;   // strictly excluded (exp -> 0)
                else if (!ok)       v = -1.0e9f;    // match reference mask fill
                sv[ct][i] = v;
            }
        }
        // prefetch next V block while we do softmax bookkeeping
        if (cc + 1 < nchunks)
            __builtin_prefetch((const void*)(Vpb + (size_t)(cc + 1) * 256 + lane * 8), 0, 0);
        // online softmax update, per row held by this half-wave
#pragma unroll
        for (int i = 0; i < 8; ++i) {
            float mx = fmaxf(sv[0][i], sv[1][i]);
#pragma unroll
            for (int d = 1; d < 16; d <<= 1) mx = fmaxf(mx, __shfl_xor(mx, d, 32));
            float nm = fmaxf(rm[i], mx);
            float f  = __expf(rm[i] - nm);
            float p0 = __expf(sv[0][i] - nm);
            float p1 = __expf(sv[1][i] - nm);
            float ps = p0 + p1;
#pragma unroll
            for (int d = 1; d < 16; d <<= 1) ps += __shfl_xor(ps, d, 32);
            rs[i] = rs[i] * f + ps;
            rm[i] = nm;
#pragma unroll
            for (int nt = 0; nt < 8; ++nt) acc[nt][i] *= f;
            ptile[(i + 8 * h) * 32 + nn]      = (bf16)p0;
            ptile[(i + 8 * h) * 32 + 16 + nn] = (bf16)p1;
        }
        __syncthreads();
        // reload probs as A-fragment (K=32), dword reads
        Frag pa;
        {
            int m = lane & 15;
#pragma unroll
            for (int i = 0; i < 4; ++i) pa.d[i]     = pt32[m * 16 + 4 * h + i];
#pragma unroll
            for (int i = 0; i < 4; ++i) pa.d[4 + i] = pt32[m * 16 + 8 + 4 * h + i];
        }
        __syncthreads();
        // acc += P @ V[c0:c0+32, :]  (V pre-packed, zero padded)
#pragma unroll
        for (int nt = 0; nt < 8; ++nt) {
            const u32* vp = Vpb + ((size_t)nt * KC + cc) * 256 + lane * 8;
            Frag vb;
#pragma unroll
            for (int i = 0; i < 8; ++i) vb.d[i] = vp[i];
            acc[nt] = __builtin_amdgcn_wmma_f32_16x16x32_bf16(false, pa.v, false, vb.v,
                                                              (short)0, acc[nt], false, false);
        }
    }
    // finalize
#pragma unroll
    for (int i = 0; i < 8; ++i) {
        int r = r0 + i + 8 * h;
        if (r < Lq) {
            float inv = 1.0f / rs[i];
#pragma unroll
            for (int nt = 0; nt < 8; ++nt)
                Ob[(size_t)r * Dm + nt * 16 + nn] = acc[nt][i] * inv;
        }
    }
}

// ---------------- pooling / gate / small GEMMs ----------------

__global__ void k_pool(const float* __restrict__ x, const float* __restrict__ pw,
                       const float* __restrict__ pb, float* __restrict__ out, int L) {
    int b = blockIdx.x;
    int d = threadIdx.x; // 128 threads
    __shared__ float x6[6][128];
    __shared__ float wgt[8];
    int cl = L / 6;
    const float* xb = x + (size_t)b * L * Dm;
    for (int j = 0; j < 6; ++j) {
        float s = 0.f;
        for (int t = 0; t < cl; ++t) s += xb[(size_t)(j * cl + t) * Dm + d];
        x6[j][d] = s / (float)cl;
    }
    __syncthreads();
    if (d < 6) {
        float s = pb[0];
        for (int k = 0; k < 128; ++k) s += x6[d][k] * pw[k];
        wgt[d] = s;
    }
    __syncthreads();
    if (d == 0) {
        float mx = wgt[0];
        for (int j = 1; j < 6; ++j) mx = fmaxf(mx, wgt[j]);
        float ss = 0.f;
        for (int j = 0; j < 6; ++j) { wgt[j] = __expf(wgt[j] - mx); ss += wgt[j]; }
        for (int j = 0; j < 6; ++j) wgt[j] /= ss;
    }
    __syncthreads();
    float o = 0.f;
    for (int j = 0; j < 6; ++j) o += wgt[j] * x6[j][d];
    out[(size_t)b * Dm + d] = o;
}

__global__ void k_sgemm(const float* __restrict__ A, const float* __restrict__ W,
                        const float* __restrict__ bias, float* __restrict__ out,
                        int M, int K, int N, int act) {
    int idx = blockIdx.x * blockDim.x + threadIdx.x;
    if (idx >= M * N) return;
    int m = idx / N, n = idx % N;
    float s = bias[n];
    const float* Ar = A + (size_t)m * K;
    for (int k = 0; k < K; ++k) s += Ar[k] * W[(size_t)k * N + n];
    if (act == 1)      s = s > 0.f ? s : 0.f;
    else if (act == 2) s = 1.0f / (1.0f + __expf(-s));
    out[idx] = s;
}

__global__ void k_concat(const float* __restrict__ a, const float* __restrict__ b,
                         float* __restrict__ z, int M, int D) {
    int idx = blockIdx.x * blockDim.x + threadIdx.x;
    if (idx >= M * 2 * D) return;
    int m = idx / (2 * D), c = idx % (2 * D);
    z[idx] = (c < D) ? a[(size_t)m * D + c] : b[(size_t)m * D + (c - D)];
}

__global__ void k_fuse(const float* __restrict__ g, const float* __restrict__ e,
                       const float* __restrict__ f, float* __restrict__ out, int n) {
    int idx = blockIdx.x * blockDim.x + threadIdx.x;
    if (idx >= n) return;
    float gv = g[idx];
    out[idx] = gv * e[idx] + (1.0f - gv) * f[idx];
}

// ---------------- orchestration ----------------

extern "C" void kernel_launch(void* const* d_in, const int* in_sizes, int n_in,
                              void* d_out, int out_size, void* d_ws, size_t ws_size,
                              hipStream_t stream) {
    (void)in_sizes; (void)n_in; (void)out_size; (void)ws_size;
    const int B = 512, TE = 600, TF = 120;
    const int ME = B * TE, MF = B * TF;
    const int KCF = 4;   // ceil(120/32)
    const int KCE = 19;  // ceil(600/32)

    const float* P[48];
    for (int i = 0; i < 48; ++i) P[i] = (const float*)d_in[i];
    const float* eeg   = P[0];
    const float* fnirs = P[1];

    char* Wp = (char*)d_ws;
    size_t off = 0;
    auto take = [&](size_t bytes) -> char* {
        char* p = Wp + off;
        off = (off + bytes + 255) & ~(size_t)255;
        return p;
    };

    // region 0: h1/h2 early, aliased by aligned_fnirs later
    float* h1 = (float*)take((size_t)B * 64 * TE * 4);
    float* h2 = (float*)take((size_t)B * 64 * TE * 4);
    float* aligned_f = h1;                               // 157 MB needed, 157 avail
    // region 1: Ae early, aliased by aligned_eeg later
    bf16* Ae = (bf16*)take((size_t)ME * 64 * 2);
    float* aligned_e = (float*)Ae;                       // 31.5 MB <= 39.3 MB
    bf16* eeg_seq = (bf16*)take((size_t)ME * 128 * 2);
    bf16* Qe = (bf16*)take((size_t)ME * 128 * 2);
    bf16* Ke = (bf16*)take((size_t)ME * 128 * 2);
    bf16* Ve = (bf16*)take((size_t)ME * 128 * 2);
    float* fdm = (float*)take((size_t)B * 72 * TF * 4);
    float* fh1 = (float*)take((size_t)B * 32 * TF * 4);
    bf16* Af = (bf16*)take((size_t)MF * 32 * 2);
    bf16* fnirs_seq = (bf16*)take((size_t)MF * 128 * 2);
    bf16* Qf = (bf16*)take((size_t)MF * 128 * 2);
    bf16* Kf = (bf16*)take((size_t)MF * 128 * 2);
    bf16* Vf = (bf16*)take((size_t)MF * 128 * 2);
    u32* VePk = (u32*)take((size_t)B * 8 * KCE * 256 * 4); // 79.7 MB
    u32* VfPk = (u32*)take((size_t)B * 8 * KCF * 256 * 4); // 16.8 MB
    float* stats1 = (float*)take(512 * 4);
    float* stats2 = (float*)take(512 * 4);
    float* statsf = (float*)take(512 * 4);
    bf16* eprojw = (bf16*)take(64 * 128 * 2);
    bf16* fprojw = (bf16*)take(32 * 128 * 2);
    bf16* qkvw[6];
    for (int i = 0; i < 6; ++i) qkvw[i] = (bf16*)take(128 * 128 * 2);
    u32* eprojp = (u32*)take((size_t)8 * 2 * 256 * 4);
    u32* fprojp = (u32*)take((size_t)8 * 1 * 256 * 4);
    u32* qkvp[6];
    for (int i = 0; i < 6; ++i) qkvp[i] = (u32*)take((size_t)8 * 4 * 256 * 4);
    float* zbuf = (float*)take((size_t)B * 256 * 4);
    float* ghid = (float*)take((size_t)B * 128 * 4);
    float* ggat = (float*)take((size_t)B * 128 * 4);

    float* out = (float*)d_out;
    float* fusion_logits = out;
    float* eeg_logits    = out + 1024;
    float* fnirs_logits  = out + 2048;
    float* se_out        = out + 3072;
    float* sf_out        = out + 23552;
    float* su_out        = out + 44032;
    float* eeg_embed     = out + 64512;
    float* fnirs_embed   = out + 130048;
    float* fused         = out + 195584;

    auto cdiv = [](int a, int b) { return (a + b - 1) / b; };

    // weights -> bf16 -> packed B-fragment layout
    k_f32_to_bf16<<<cdiv(64 * 128, 256), 256, 0, stream>>>(P[10], eprojw, 64 * 128);
    k_f32_to_bf16<<<cdiv(32 * 128, 256), 256, 0, stream>>>(P[16], fprojw, 32 * 128);
    for (int i = 0; i < 6; ++i)
        k_f32_to_bf16<<<cdiv(128 * 128, 256), 256, 0, stream>>>(P[18 + 2 * i], qkvw[i], 128 * 128);
    k_packB<<<cdiv(8 * 2 * 256, 256), 256, 0, stream>>>(eprojw, eprojp, 64, 2, 1);
    k_packB<<<cdiv(8 * 1 * 256, 256), 256, 0, stream>>>(fprojw, fprojp, 32, 1, 1);
    for (int i = 0; i < 6; ++i)
        k_packB<<<cdiv(8 * 4 * 256, 256), 256, 0, stream>>>(qkvw[i], qkvp[i], 128, 4, 1);

    // EEG encoder
    k_conv1d<<<cdiv(B * 64 * TE, 256), 256, 0, stream>>>(eeg, P[2], P[3], h1, B, 28, 64, TE, 7);
    k_bnstats<<<64, 256, 0, stream>>>(h1, stats1, B, 64, TE);
    k_bn_relu<<<cdiv(B * 64 * TE, 256), 256, 0, stream>>>(h1, stats1, P[4], P[5], B, 64, TE);
    k_conv1d<<<cdiv(B * 64 * TE, 256), 256, 0, stream>>>(h1, P[6], P[7], h2, B, 64, 64, TE, 5);
    k_bnstats<<<64, 256, 0, stream>>>(h2, stats2, B, 64, TE);
    k_bn_relu<<<cdiv(B * 64 * TE, 256), 256, 0, stream>>>(h2, stats2, P[8], P[9], B, 64, TE);
    k_tc<<<cdiv(ME * 64, 256), 256, 0, stream>>>(h2, Ae, B, 64, TE);
    k_gemm128<<<cdiv((ME / 16) * 2, 8), 256, 0, stream>>>(Ae, eprojp, P[11], eeg_seq, ME, 64);

    // fNIRS encoder
    k_demean<<<cdiv(B * 72, 256), 256, 0, stream>>>(fnirs, fdm, B * 72, TF);
    k_conv1d<<<cdiv(B * 32 * TF, 256), 256, 0, stream>>>(fdm, P[12], P[13], fh1, B, 72, 32, TF, 5);
    k_bnstats<<<32, 256, 0, stream>>>(fh1, statsf, B, 32, TF);
    k_bn_relu<<<cdiv(B * 32 * TF, 256), 256, 0, stream>>>(fh1, statsf, P[14], P[15], B, 32, TF);
    k_tc<<<cdiv(MF * 32, 256), 256, 0, stream>>>(fh1, Af, B, 32, TF);
    k_gemm128<<<cdiv((MF / 16) * 2, 8), 256, 0, stream>>>(Af, fprojp, P[17], fnirs_seq, MF, 32);

    // QKV projections
    bf16* qkvo_e[3] = {Qe, Ke, Ve};
    for (int i = 0; i < 3; ++i)
        k_gemm128<<<cdiv((ME / 16) * 2, 8), 256, 0, stream>>>(eeg_seq, qkvp[i], P[19 + 2 * i],
                                                              qkvo_e[i], ME, 128);
    bf16* qkvo_f[3] = {Qf, Kf, Vf};
    for (int i = 0; i < 3; ++i)
        k_gemm128<<<cdiv((MF / 16) * 2, 8), 256, 0, stream>>>(fnirs_seq, qkvp[3 + i], P[25 + 2 * i],
                                                              qkvo_f[i], MF, 128);

    // pack V matrices into B-fragment layout (zero padded past Lk)
    k_packB<<<cdiv(B * 8 * KCF * 256, 256), 256, 0, stream>>>(Vf, VfPk, TF, KCF, B);
    k_packB<<<cdiv(B * 8 * KCE * 256, 256), 256, 0, stream>>>(Ve, VePk, TE, KCE, B);

    // banded cross attention (flash style, per batch)
    int tilesE = cdiv(TE, 16), tilesF = cdiv(TF, 16);
    k_attn<<<B * tilesE, 32, 0, stream>>>(Qe, Kf, VfPk, aligned_f, TE, TF, 0); // aligned_fnirs
    k_attn<<<B * tilesF, 32, 0, stream>>>(Qf, Ke, VePk, aligned_e, TF, TE, 1); // aligned_eeg

    // attention pooling (note: swapped assignment per reference)
    k_pool<<<B, 128, 0, stream>>>(aligned_f, P[30], P[31], eeg_embed, TE);
    k_pool<<<B, 128, 0, stream>>>(aligned_e, P[30], P[31], fnirs_embed, TF);

    // heads
    k_sgemm<<<cdiv(B * 2, 256), 256, 0, stream>>>(eeg_embed, P[38], P[39], eeg_logits, B, 128, 2, 0);
    k_sgemm<<<cdiv(B * 2, 256), 256, 0, stream>>>(fnirs_embed, P[40], P[41], fnirs_logits, B, 128, 2, 0);
    k_sgemm<<<cdiv(B * 40, 256), 256, 0, stream>>>(eeg_embed, P[42], P[43], se_out, B, 128, 40, 0);
    k_sgemm<<<cdiv(B * 40, 256), 256, 0, stream>>>(fnirs_embed, P[44], P[45], sf_out, B, 128, 40, 0);

    // gated fusion
    k_concat<<<cdiv(B * 256, 256), 256, 0, stream>>>(eeg_embed, fnirs_embed, zbuf, B, 128);
    k_sgemm<<<cdiv(B * 128, 256), 256, 0, stream>>>(zbuf, P[32], P[33], ghid, B, 256, 128, 1);
    k_sgemm<<<cdiv(B * 128, 256), 256, 0, stream>>>(ghid, P[34], P[35], ggat, B, 128, 128, 2);
    k_fuse<<<cdiv(B * 128, 256), 256, 0, stream>>>(ggat, eeg_embed, fnirs_embed, fused, B * 128);
    k_sgemm<<<cdiv(B * 2, 256), 256, 0, stream>>>(fused, P[36], P[37], fusion_logits, B, 128, 2, 0);
    k_sgemm<<<cdiv(B * 40, 256), 256, 0, stream>>>(fused, P[46], P[47], su_out, B, 128, 40, 0);
}
